// ManyWorldsEngine_12395275616279
// MI455X (gfx1250) — matrix operand order repaired and checked by reading, not compiled, not run on hardware
//
#include <hip/hip_runtime.h>
#include <hip/hip_bf16.h>
#include <math.h>

#define NB 8
#define NC 16384
#define HID 128

typedef __attribute__((ext_vector_type(16))) __bf16 v16bf;
typedef __attribute__((ext_vector_type(8)))  __bf16 v8bf;
typedef __attribute__((ext_vector_type(8)))  float  v8f;

// ---------- workspace layout (bytes) ----------
// bf16 region:
//   WaT   @ 0       (128K x 128N)   32768 B   (eaW1[:,64:] transposed K-major)
//   WgT   @ 32768   (128 x 128)     32768 B
//   W2aT  @ 65536   (128K x 64N)    16384 B
//   W2gT  @ 81920                   16384 B
//   WihT  @ 98304   (64K x 384N)    49152 B   (gru_Wih[:, :64] K-major)
//   WhhT  @ 147456  (128K x 384N)   98304 B
// f32 region @ 245760 (float index offsets):
//   ca 0(1024) cg 1024(1024) dB2 2048(64) aabs 2112(8)
//   fsum 2120(8192) fdcsum 10312(8192) A1 18504(8192) A2 26696(1024) If 27720(128)
//   cellout 27848(1048576) avgt 1076424(16384)
#define F32_OFF 245760

static __device__ __forceinline__ v16bf cat16(v8bf lo, v8bf hi) {
  return __builtin_shufflevector(lo, hi, 0,1,2,3,4,5,6,7,8,9,10,11,12,13,14,15);
}

// A fragment (16x32 bf16) from LDS row-major [16 x Kd]; lane = M, two 8-elem runs.
static __device__ __forceinline__ v16bf load_a(const __bf16* buf, int Kd, int kt, int row, int half) {
  const __bf16* p = buf + row * Kd + kt * 32 + half * 8;
  v8bf lo = *(const v8bf*)p;        // K = base..base+7    -> elements 0..7
  v8bf hi = *(const v8bf*)(p + 16); // K = base+16..+23    -> elements 8..15
  return cat16(lo, hi);
}

// B fragment (32x16 bf16) from global K-major weights [K x Nd]; lane = N (lane&15), elem = K (table order).
static __device__ __forceinline__ v16bf load_b(const __bf16* W, int Nd, int kbase, int nbase, int lane) {
  const __bf16* p = W + (size_t)(kbase + (lane >> 4) * 8) * Nd + nbase + (lane & 15);
  v16bf r;
#pragma unroll
  for (int s = 0; s < 8; ++s) {
    r[s]     = p[(size_t)s * Nd];
    r[s + 8] = p[(size_t)(s + 16) * Nd];
  }
  return r;
}

#define WMMA_BF16(A, B, C) \
  __builtin_amdgcn_wmma_f32_16x16x32_bf16(false, (A), false, (B), (short)0, (C), false, false)

static __device__ __forceinline__ float fast_sigmoid(float x) {
  return 1.f / (1.f + __expf(-x));
}
static __device__ __forceinline__ float fast_tanh(float x) {
  x = fminf(fmaxf(x, -15.f), 15.f);
  float e2 = __expf(2.f * x);
  return (e2 - 1.f) / (e2 + 1.f);
}

// ---------------- setup 0: weight repack to bf16 K-major + zero accumulators ----
__global__ void k_setup0(const float* __restrict__ eaW1, const float* __restrict__ egW1,
                         const float* __restrict__ eaW2, const float* __restrict__ egW2,
                         const float* __restrict__ gWih, const float* __restrict__ gWhh,
                         char* __restrict__ ws) {
  __bf16* WaT  = (__bf16*)(ws);
  __bf16* WgT  = (__bf16*)(ws + 32768);
  __bf16* W2aT = (__bf16*)(ws + 65536);
  __bf16* W2gT = (__bf16*)(ws + 81920);
  __bf16* WihT = (__bf16*)(ws + 98304);
  __bf16* WhhT = (__bf16*)(ws + 147456);
  float*  f32  = (float*)(ws + F32_OFF);
  float*  fsum = f32 + 2120;                  // fsum+fdcsum contiguous: 16384 floats
  float*  acc0 = f32 + 27848;                 // cellout+avgt contiguous: 1064960 floats
  int tid = blockIdx.x * blockDim.x + threadIdx.x;
  int stride = gridDim.x * blockDim.x;
  for (int i = tid; i < 128 * 128; i += stride) {
    int k = i >> 7, e = i & 127;
    WaT[i] = (__bf16)eaW1[e * 192 + 64 + k];
    WgT[i] = (__bf16)egW1[e * 192 + 64 + k];
  }
  for (int i = tid; i < 128 * 64; i += stride) {
    int k = i >> 6, o = i & 63;
    W2aT[i] = (__bf16)eaW2[o * 128 + k];
    W2gT[i] = (__bf16)egW2[o * 128 + k];
  }
  for (int i = tid; i < 64 * 384; i += stride) {
    int k = i / 384, j = i % 384;
    WihT[i] = (__bf16)gWih[j * 65 + k];
  }
  for (int i = tid; i < 128 * 384; i += stride) {
    int k = i / 384, j = i % 384;
    WhhT[i] = (__bf16)gWhh[j * 128 + k];
  }
  for (int i = tid; i < 16384; i += stride) fsum[i] = 0.f;
  for (int i = tid; i < 1064960; i += stride) acc0[i] = 0.f;
}

// ---------------- setup 1: fold constant input part into per-branch biases -------
__global__ void k_setup1(const float* __restrict__ x, const float* __restrict__ noise,
                         const float* __restrict__ amps,
                         const float* __restrict__ eaW1, const float* __restrict__ eaB1,
                         const float* __restrict__ egW1, const float* __restrict__ egB1,
                         const float* __restrict__ eaB2, const float* __restrict__ egB2,
                         char* __restrict__ ws) {
  float* f32 = (float*)(ws + F32_OFF);
  float* ca = f32, *cg = f32 + 1024, *dB2 = f32 + 2048, *aabs = f32 + 2112;
  int t = threadIdx.x;
  for (int id = t; id < 2048; id += blockDim.x) {
    int which = id >> 10, b = (id >> 7) & 7, e = id & 127;
    const float* W  = which ? egW1 : eaW1;
    const float* Bv = which ? egB1 : eaB1;
    float sc = 0.05f * (float)(b + 1);
    float s = Bv[e];
    for (int k = 0; k < 64; ++k) s += (x[k] + noise[b * 64 + k] * sc) * W[e * 192 + k];
    (which ? cg : ca)[b * 128 + e] = s;
  }
  if (t < 64) dB2[t] = eaB2[t] - egB2[t];
  if (t == 0) {
    float s = 1e-8f;
    for (int b = 0; b < 8; ++b) s += fabsf(amps[b]);
    for (int b = 0; b < 8; ++b) aabs[b] = fabsf(amps[b]) / s;
  }
}

// ---------------- main WMMA kernel: per-cell MLPs + GRU, stats accumulation -----
// grid: (NC/64, NB); block: 128 threads = 4 waves x 16 cells.
__global__ void __launch_bounds__(128) k_main(const float* __restrict__ hiddens,
                                              const float* __restrict__ gWih,
                                              const float* __restrict__ gbih,
                                              const float* __restrict__ gbhh,
                                              char* __restrict__ ws,
                                              float* __restrict__ newh_out) {
  __shared__ __align__(16) __bf16 sm[4 * 5120]; // per wave: hB 16x128, tB 16x128, oB 16x64
  const int lane = threadIdx.x & 31;
  const int wv   = threadIdx.x >> 5;
  const int row  = lane & 15;   // A-row == C-column (lane&15)
  const int half = lane >> 4;
  const int cell0 = blockIdx.x * 64 + wv * 16;
  const int b     = blockIdx.y;

  __bf16* hB = sm + wv * 5120;
  __bf16* tB = hB + 2048;
  __bf16* oB = tB + 2048;

  const __bf16* WaT  = (const __bf16*)(ws);
  const __bf16* WgT  = (const __bf16*)(ws + 32768);
  const __bf16* W2aT = (const __bf16*)(ws + 65536);
  const __bf16* W2gT = (const __bf16*)(ws + 81920);
  const __bf16* WihT = (const __bf16*)(ws + 98304);
  const __bf16* WhhT = (const __bf16*)(ws + 147456);
  float* f32 = (float*)(ws + F32_OFF);
  const float* ca = f32;
  const float* cg = f32 + 1024;
  const float* dB2 = f32 + 2048;
  const float* aabs = f32 + 2112;
  float* fsum    = f32 + 2120;
  float* fdcsum  = f32 + 10312;
  float* cellout = f32 + 27848;
  float* avgt    = f32 + 1076424;

  const int  fac = cell0 >> 11;              // faction (2048 cells each)
  const bool dcf = (cell0 & 2047) < 512;     // debate core region

  // stage h tile (f32 -> bf16 LDS); wave-local, DS in-order: no barriers needed
  const float* hsrc = hiddens + (size_t)(b * NC + cell0) * HID;
  const float4* hsrc4 = (const float4*)hsrc;
  for (int i = lane; i < 512; i += 32) {
    float4 v = hsrc4[i];
    hB[4 * i + 0] = (__bf16)v.x; hB[4 * i + 1] = (__bf16)v.y;
    hB[4 * i + 2] = (__bf16)v.z; hB[4 * i + 3] = (__bf16)v.w;
  }
  v16bf ka[4];
#pragma unroll
  for (int kt = 0; kt < 4; ++kt) ka[kt] = load_a(hB, 128, kt, row, half);

  // ha = relu(h @ Wa^T + ca[b])  -> tB
#pragma unroll
  for (int nt = 0; nt < 8; ++nt) {
    float bias = ca[b * 128 + nt * 16 + row];
    v8f acc; for (int i = 0; i < 8; ++i) acc[i] = bias;
#pragma unroll
    for (int kt = 0; kt < 4; ++kt)
      acc = WMMA_BF16(ka[kt], load_b(WaT, 128, kt * 32, nt * 16, lane), acc);
#pragma unroll
    for (int i = 0; i < 8; ++i)
      tB[(i + 8 * half) * 128 + nt * 16 + row] = (__bf16)fmaxf(acc[i], 0.f);
  }
  // a = ha @ W2a^T + (eaB2-egB2)
  v16bf ta[4];
#pragma unroll
  for (int kt = 0; kt < 4; ++kt) ta[kt] = load_a(tB, 128, kt, row, half);
  v8f av[4];
#pragma unroll
  for (int nt = 0; nt < 4; ++nt) {
    float bias = dB2[nt * 16 + row];
    v8f acc; for (int i = 0; i < 8; ++i) acc[i] = bias;
#pragma unroll
    for (int kt = 0; kt < 4; ++kt)
      acc = WMMA_BF16(ta[kt], load_b(W2aT, 64, kt * 32, nt * 16, lane), acc);
    av[nt] = acc;
  }
  // hg = relu(h @ Wg^T + cg[b]) -> tB (reuse)
#pragma unroll
  for (int nt = 0; nt < 8; ++nt) {
    float bias = cg[b * 128 + nt * 16 + row];
    v8f acc; for (int i = 0; i < 8; ++i) acc[i] = bias;
#pragma unroll
    for (int kt = 0; kt < 4; ++kt)
      acc = WMMA_BF16(ka[kt], load_b(WgT, 128, kt * 32, nt * 16, lane), acc);
#pragma unroll
    for (int i = 0; i < 8; ++i)
      tB[(i + 8 * half) * 128 + nt * 16 + row] = (__bf16)fmaxf(acc[i], 0.f);
  }
  // out = a - hg @ W2g^T
#pragma unroll
  for (int kt = 0; kt < 4; ++kt) ta[kt] = load_a(tB, 128, kt, row, half);
  v8f ov[4];
#pragma unroll
  for (int nt = 0; nt < 4; ++nt) {
    v8f acc; for (int i = 0; i < 8; ++i) acc[i] = 0.f;
#pragma unroll
    for (int kt = 0; kt < 4; ++kt)
      acc = WMMA_BF16(ta[kt], load_b(W2gT, 64, kt * 32, nt * 16, lane), acc);
    ov[nt] = av[nt] - acc;
#pragma unroll
    for (int i = 0; i < 8; ++i)
      oB[(i + 8 * half) * 64 + nt * 16 + row] = (__bf16)ov[nt][i];
  }
  // tension = mean(out^2) per row (butterfly over the 16-lane half-group)
  float ten[8];
#pragma unroll
  for (int i = 0; i < 8; ++i) {
    float s = 0.f;
#pragma unroll
    for (int nt = 0; nt < 4; ++nt) s += ov[nt][i] * ov[nt][i];
    ten[i] = s;
  }
#pragma unroll
  for (int m = 1; m <= 8; m <<= 1)
#pragma unroll
    for (int i = 0; i < 8; ++i) ten[i] += __shfl_xor(ten[i], m, 32);
#pragma unroll
  for (int i = 0; i < 8; ++i) ten[i] *= (1.f / 64.f);
  // cross-branch accumulators (disjoint per (cell,col) x 8 branches -> atomics)
  {
    float ab = aabs[b];
#pragma unroll
    for (int nt = 0; nt < 4; ++nt)
#pragma unroll
      for (int i = 0; i < 8; ++i)
        atomicAdd(&cellout[(size_t)(cell0 + i + 8 * half) * 64 + nt * 16 + row],
                  ab * ov[nt][i]);
    if (row == 0)
#pragma unroll
      for (int i = 0; i < 8; ++i)
        atomicAdd(&avgt[cell0 + i + 8 * half], ten[i] * 0.125f);
  }

  // GRU gates, one 16-column slice at a time (r,z,inn,hn accumulators live)
  v16bf oa[2];
  oa[0] = load_a(oB, 64, 0, row, half);
  oa[1] = load_a(oB, 64, 1, row, half);
  for (int nt = 0; nt < 8; ++nt) {
    int c = nt * 16 + row;
    float wtr = gWih[(0   + c) * 65 + 64];
    float wtz = gWih[(128 + c) * 65 + 64];
    float wtn = gWih[(256 + c) * 65 + 64];
    float br  = gbih[c]       + gbhh[c];
    float bz  = gbih[128 + c] + gbhh[128 + c];
    float bin = gbih[256 + c];
    float bhn = gbhh[256 + c];
    v8f ar, az, ai, ah;
    for (int i = 0; i < 8; ++i) {
      ar[i] = br  + ten[i] * wtr;
      az[i] = bz  + ten[i] * wtz;
      ai[i] = bin + ten[i] * wtn;
      ah[i] = bhn;
    }
    ar = WMMA_BF16(oa[0], load_b(WihT, 384, 0,  nt * 16,        lane), ar);
    ar = WMMA_BF16(oa[1], load_b(WihT, 384, 32, nt * 16,        lane), ar);
    az = WMMA_BF16(oa[0], load_b(WihT, 384, 0,  128 + nt * 16,  lane), az);
    az = WMMA_BF16(oa[1], load_b(WihT, 384, 32, 128 + nt * 16,  lane), az);
    ai = WMMA_BF16(oa[0], load_b(WihT, 384, 0,  256 + nt * 16,  lane), ai);
    ai = WMMA_BF16(oa[1], load_b(WihT, 384, 32, 256 + nt * 16,  lane), ai);
#pragma unroll
    for (int kt = 0; kt < 4; ++kt) {
      ar = WMMA_BF16(ka[kt], load_b(WhhT, 384, kt * 32, nt * 16,       lane), ar);
      az = WMMA_BF16(ka[kt], load_b(WhhT, 384, kt * 32, 128 + nt * 16, lane), az);
      ah = WMMA_BF16(ka[kt], load_b(WhhT, 384, kt * 32, 256 + nt * 16, lane), ah);
    }
    float rs = 0.f;
#pragma unroll
    for (int i = 0; i < 8; ++i) {
      float r  = fast_sigmoid(ar[i]);
      float z  = fast_sigmoid(az[i]);
      float n  = fast_tanh(ai[i] + r * ah[i]);
      float hv = (float)hB[(i + 8 * half) * 128 + c];
      float nh = (1.f - z) * n + z * hv;
      newh_out[(size_t)(b * NC + cell0 + i + 8 * half) * HID + c] = nh;
      rs += nh;
    }
    rs += __shfl_xor(rs, 16, 32);
    if (half == 0) {
      atomicAdd(&fsum[(b * 8 + fac) * 128 + c], rs);
      if (dcf) atomicAdd(&fdcsum[(b * 8 + fac) * 128 + c], rs);
    }
  }
}

// ---------------- analytic sync/debate terms + mixer interference ---------------
__global__ void __launch_bounds__(1024) k_sync(char* __restrict__ ws,
                                               const float* __restrict__ mixW,
                                               const float* __restrict__ mixb,
                                               const int* __restrict__ stepp) {
  __shared__ float mu[1024];
  float* f32 = (float*)(ws + F32_OFF);
  const float* fsum   = f32 + 2120;
  const float* fdcsum = f32 + 10312;
  float* A1 = f32 + 18504;
  float* A2 = f32 + 26696;
  float* If = f32 + 27720;
  int t = threadIdx.x, b = t >> 7, h = t & 127;
  float dbt = (*stepp > 5) ? 0.15f : 0.f;
  float S[8], tot = 0.f;
  for (int f = 0; f < 8; ++f) { S[f] = fsum[(b * 8 + f) * 128 + h]; tot += S[f]; }
  float glob = tot * (1.f / 16384.f);
  float after = 0.f;
  for (int f = 0; f < 8; ++f) {
    float fm = S[f] * (1.f / 2048.f);
    A1[(b * 8 + f) * 128 + h] = 0.15f * fm;
    float Sd = fdcsum[(b * 8 + f) * 128 + h];
    // sum after sync (=S[f]) plus debate correction on the first 512 cells
    after += S[f] + dbt * (512.f * glob - (0.85f * Sd + 0.15f * 512.f * fm));
  }
  A2[b * 128 + h] = 0.15f * glob;
  mu[t] = after * (1.f / 16384.f);   // mean of newh after sync+debate, pre-interf
  __syncthreads();
  if (t < 128) {
    float s = mixb[t];
    for (int j = 0; j < 1024; ++j) s += mu[j] * mixW[t * 1024 + j];
    If[t] = 0.05f * s;
  }
}

// ---------------- apply sync/debate/interference to newh in place ---------------
__global__ void k_apply(float* __restrict__ newh, char* __restrict__ ws,
                        const int* __restrict__ stepp) {
  float* f32 = (float*)(ws + F32_OFF);
  const float* A1 = f32 + 18504;
  const float* A2 = f32 + 26696;
  const float* If = f32 + 27720;
  int step = *stepp;
  size_t idx = (size_t)blockIdx.x * blockDim.x + threadIdx.x;
  const size_t total = (size_t)NB * NC * HID;
  const size_t stride = (size_t)gridDim.x * blockDim.x;
  for (; idx < total; idx += stride) {
    int c = (int)(idx & 127);
    int n = (int)((idx >> 7) & 16383);
    int b = (int)(idx >> 21);
    int f = n >> 11, pos = n & 2047;
    float v = newh[idx];
    v = 0.85f * v + A1[(b * 8 + f) * 128 + c];
    if (step > 5 && pos < 512) v = 0.85f * v + A2[b * 128 + c];
    if (b == 0) v += If[c];
    newh[idx] = v;
  }
}

// ---------------- softmax over tensions + head projection + scalar --------------
__global__ void __launch_bounds__(1024) k_head(char* __restrict__ ws,
                                               const float* __restrict__ headW,
                                               const float* __restrict__ headb,
                                               float* __restrict__ outp) {
  float* f32 = (float*)(ws + F32_OFF);
  const float* cellout = f32 + 27848;
  const float* avgt    = f32 + 1076424;
  __shared__ float red[1024];
  __shared__ float cred[64];
  __shared__ float comb[64];
  int t = threadIdx.x;
  float a[16];
  float mx = -3.4e38f, ts = 0.f;
#pragma unroll
  for (int j = 0; j < 16; ++j) { a[j] = avgt[j * 1024 + t]; mx = fmaxf(mx, a[j]); ts += a[j]; }
  red[t] = mx; __syncthreads();
  for (int s = 512; s > 0; s >>= 1) { if (t < s) red[t] = fmaxf(red[t], red[t + s]); __syncthreads(); }
  mx = red[0]; __syncthreads();
  float w[16], se = 0.f;
#pragma unroll
  for (int j = 0; j < 16; ++j) { w[j] = __expf(a[j] - mx); se += w[j]; }
  red[t] = se; __syncthreads();
  for (int s = 512; s > 0; s >>= 1) { if (t < s) red[t] += red[t + s]; __syncthreads(); }
  se = red[0]; __syncthreads();
  red[t] = ts; __syncthreads();
  for (int s = 512; s > 0; s >>= 1) { if (t < s) red[t] += red[t + s]; __syncthreads(); }
  float tstot = red[0];
  if (t < 64) cred[t] = 0.f;
  __syncthreads();
  float part[64];
#pragma unroll
  for (int o = 0; o < 64; ++o) part[o] = 0.f;
  for (int j = 0; j < 16; ++j) {
    const float* cp = cellout + (size_t)(j * 1024 + t) * 64;
#pragma unroll
    for (int o = 0; o < 64; ++o) part[o] += w[j] * cp[o];
  }
#pragma unroll
  for (int o = 0; o < 64; ++o) atomicAdd(&cred[o], part[o]);
  __syncthreads();
  if (t < 64) comb[t] = cred[t] / se;
  __syncthreads();
  if (t < 64) {
    float p = headb[t];
#pragma unroll
    for (int o = 0; o < 64; ++o) p += comb[o] * headW[t * 64 + o];
    outp[t] = p;
  }
  if (t == 0) outp[64] = tstot * (1.f / 16384.f);
}

extern "C" void kernel_launch(void* const* d_in, const int* in_sizes, int n_in,
                              void* d_out, int out_size, void* d_ws, size_t ws_size,
                              hipStream_t stream) {
  const float* x       = (const float*)d_in[0];
  const float* noise   = (const float*)d_in[1];
  const float* amps    = (const float*)d_in[2];
  const float* hiddens = (const float*)d_in[3];
  const float* eaW1    = (const float*)d_in[4];
  const float* eaB1    = (const float*)d_in[5];
  const float* eaW2    = (const float*)d_in[6];
  const float* eaB2    = (const float*)d_in[7];
  const float* egW1    = (const float*)d_in[8];
  const float* egB1    = (const float*)d_in[9];
  const float* egW2    = (const float*)d_in[10];
  const float* egB2    = (const float*)d_in[11];
  const float* gWih    = (const float*)d_in[12];
  const float* gWhh    = (const float*)d_in[13];
  const float* gbih    = (const float*)d_in[14];
  const float* gbhh    = (const float*)d_in[15];
  const float* headW   = (const float*)d_in[16];
  const float* headb   = (const float*)d_in[17];
  const float* mixW    = (const float*)d_in[18];
  const float* mixb    = (const float*)d_in[19];
  const int*   step    = (const int*)d_in[20];
  float* out  = (float*)d_out;
  float* newh = out + 65;              // (pred 64) + (scalar 1) + newh
  char*  ws   = (char*)d_ws;

  k_setup0<<<1024, 256, 0, stream>>>(eaW1, egW1, eaW2, egW2, gWih, gWhh, ws);
  k_setup1<<<1, 256, 0, stream>>>(x, noise, amps, eaW1, eaB1, egW1, egB1, eaB2, egB2, ws);
  dim3 grid(NC / 64, NB);
  k_main<<<grid, 128, 0, stream>>>(hiddens, gWih, gbih, gbhh, ws, newh);
  k_sync<<<1, 1024, 0, stream>>>(ws, mixW, mixb, step);
  k_apply<<<8192, 256, 0, stream>>>(newh, ws, step);
  k_head<<<1, 1024, 0, stream>>>(ws, headW, headb, out);
}
